// AnchorDataGenerator_22728966930550
// MI455X (gfx1250) — compile-verified, matrix operand order checked by reading, not run on hardware
//
#include <hip/hip_runtime.h>
#include <hip/hip_bf16.h>

// ---------------------------------------------------------------------------
// RPN anchor-target generator for MI455X (gfx1250, wave32).
// H=W=256, A=9 anchors/cell, N=589824 anchors, G=64 gt boxes.
//   k_init   : zero the 64 per-gt max keys
//   k_gmax   : per-gt max IoU over valid anchors; gt tile staged into LDS via
//              the Tensor Data Mover (tensor_load_to_lds + s_wait_tensorcnt)
//   k_labels : labels/argmax; fg/bg population counts via V_WMMA_F32_16X16X32_F16
//              (ones-matmul reduction); gt staged via global_load_async_to_lds
//   k_scan   : tiny exclusive scan of per-block counts; num_ni
//   k_out    : deterministic first-128 subsample via wave32 ballot ranks,
//              bbox targets, weights, NCHW-transposed writes
// ---------------------------------------------------------------------------

#define HH 256
#define WW 256
#define NA 9
#define NG 64
#define NANCH (HH * WW * NA)       // 589824
#define LBL_BLOCKS 144             // NANCH / 4096
#define CAP 128

typedef __attribute__((ext_vector_type(16))) _Float16 v16h;
typedef __attribute__((ext_vector_type(8)))  float    v8f;

typedef unsigned int u32x4 __attribute__((ext_vector_type(4)));
typedef int          i32x4 __attribute__((ext_vector_type(4)));
typedef int          i32x8 __attribute__((ext_vector_type(8)));

typedef __attribute__((address_space(1))) int  gas_i32;   // global-AS int
typedef __attribute__((address_space(3))) int  las_i32;   // LDS-AS int
typedef __attribute__((address_space(3))) float las_f32;  // LDS-AS float

#if defined(__gfx1250__) && __has_builtin(__builtin_amdgcn_tensor_load_to_lds) && \
    __has_builtin(__builtin_amdgcn_s_wait_tensorcnt)
#define HAVE_TDM 1
#else
#define HAVE_TDM 0
#endif

#if defined(__gfx1250__) && __has_builtin(__builtin_amdgcn_global_load_async_to_lds_b32)
#define HAVE_ASYNC 1
#else
#define HAVE_ASYNC 0
#endif

// Base anchor W/H per anchor index a = ri*3+si  (ratios 0.5,1,2 ; scales 8,16,32)
// ws_base={23,16,11}, hs_base={12,16,22} (np.round incl. banker's 11.5->12)
__constant__ float c_AW[9] = {184.f, 368.f, 736.f, 128.f, 256.f, 512.f,  88.f, 176.f, 352.f};
__constant__ float c_AH[9] = { 96.f, 192.f, 384.f, 128.f, 256.f, 512.f, 176.f, 352.f, 704.f};

__device__ __forceinline__ void anchor_coords(int i, float& x1, float& y1,
                                              float& x2, float& y2) {
  int a = i % NA;
  int pos = i / NA;
  int w = pos % WW;
  int h = pos / WW;
  float Wd = c_AW[a], Hd = c_AH[a];
  float sx = (float)(w << 4), sy = (float)(h << 4);   // stride 16
  x1 = sx + 7.5f - 0.5f * (Wd - 1.f);
  y1 = sy + 7.5f - 0.5f * (Hd - 1.f);
  x2 = sx + 7.5f + 0.5f * (Wd - 1.f);
  y2 = sy + 7.5f + 0.5f * (Hd - 1.f);
}

__device__ __forceinline__ float iou_box(float ax1, float ay1, float ax2, float ay2,
                                         float aarea,
                                         float gx1, float gy1, float gx2, float gy2,
                                         float garea) {
  float ix1 = fmaxf(ax1, gx1), iy1 = fmaxf(ay1, gy1);
  float ix2 = fminf(ax2, gx2), iy2 = fminf(ay2, gy2);
  float iw = ix2 - ix1 + 1.f, ih = iy2 - iy1 + 1.f;
  float inter = (iw > 0.f && ih > 0.f) ? iw * ih : 0.f;
  return inter / (aarea + garea - inter);
}

// Order-preserving float <-> uint key (for atomicMax over floats incl. -1)
__device__ __forceinline__ unsigned f2key(float f) {
  unsigned u = __float_as_uint(f);
  return (u & 0x80000000u) ? ~u : (u | 0x80000000u);
}
__device__ __forceinline__ float key2f(unsigned k) {
  unsigned u = (k & 0x80000000u) ? (k & 0x7fffffffu) : ~k;
  return __uint_as_float(u);
}

// Stage the 64x4 f32 gt table (256 dwords) into LDS via async-to-LDS loads
// (ASYNCcnt-tracked).  Threads t < 256 participate; caller must
// __syncthreads() after return.
__device__ __forceinline__ void stage_gt_async(const float* __restrict__ gt,
                                               float* sg, int t) {
#if HAVE_ASYNC
  if (t < NG * 4) {
    __builtin_amdgcn_global_load_async_to_lds_b32(
        (gas_i32*)(gt + t), (las_i32*)&sg[t], 0, 0);
  }
#if __has_builtin(__builtin_amdgcn_s_wait_asynccnt)
  __builtin_amdgcn_s_wait_asynccnt(0);
#else
  asm volatile("s_wait_asynccnt 0" ::: "memory");
#endif
#else
  if (t < NG * 4) sg[t] = gt[t];
#endif
}

__global__ void k_init(unsigned* gmaxKey) {
  if (threadIdx.x < NG) gmaxKey[threadIdx.x] = 0u;   // key 0 == smallest
}

// ---------------------------------------------------------------------------
// Per-gt max over valid anchors.  Block = 256 threads covers 256 anchors.
// gt tile staged by the Tensor Data Mover: one tensor_load_to_lds from wave 0,
// tracked by TENSORcnt (s_wait_tensorcnt), then published via barrier.
// ---------------------------------------------------------------------------
__global__ void k_gmax(const float* __restrict__ gt, const int* __restrict__ imw,
                       const int* __restrict__ imh, unsigned* __restrict__ gmaxKey) {
  __shared__ float sg[NG * 4];
  __shared__ float sga[NG];
  __shared__ unsigned skey[NG];
  int t = threadIdx.x;

#if HAVE_TDM
  if ((t >> 5) == 0) {                       // one TDM op per block (wave 0)
    unsigned long long ga = (unsigned long long)gt;
    unsigned lds_off = (unsigned)(unsigned long long)(las_f32*)&sg[0];
    u32x4 g0;
    g0[0] = 1u;                                             // count=1, user D#
    g0[1] = lds_off;                                        // lds_addr (bytes)
    g0[2] = (unsigned)(ga & 0xffffffffull);                 // global_addr[31:0]
    g0[3] = (unsigned)((ga >> 32) & 0x01ffffffull)          // global_addr[56:32]
            | (2u << 30);                                   // type = 2 ("image")
    i32x8 g1;
    g1[0] = (int)(2u << 16);          // workgroup_mask=0, data_size=2 (4 bytes)
    g1[1] = (int)(256u << 16);        // tensor_dim0[15:0] = 256 (bits 63:48)
    g1[2] = (int)(1u << 16);          // tensor_dim0 hi = 0, tensor_dim1 = 1
    g1[3] = (int)(256u << 16);        // tile_dim0 = 256 (bits 127:112)
    g1[4] = 1;                        // tile_dim1 = 1, tile_dim2 = 0
    g1[5] = 256;                      // tensor_dim0_stride = 256
    g1[6] = 0;
    g1[7] = 0;
    i32x4 g2; g2[0] = 0; g2[1] = 0; g2[2] = 0; g2[3] = 0;
    i32x4 g3; g3[0] = 0; g3[1] = 0; g3[2] = 0; g3[3] = 0;
#if __clang_major__ >= 23
    i32x8 g4;
    g4[0] = 0; g4[1] = 0; g4[2] = 0; g4[3] = 0;
    g4[4] = 0; g4[5] = 0; g4[6] = 0; g4[7] = 0;
    __builtin_amdgcn_tensor_load_to_lds(g0, g1, g2, g3, g4, 0);
#else
    __builtin_amdgcn_tensor_load_to_lds(g0, g1, g2, g3, 0);
#endif
    __builtin_amdgcn_s_wait_tensorcnt(0);
  }
#else
  if (t < NG * 4) sg[t] = gt[t];
#endif
  __syncthreads();

  if (t < NG) {
    float gw = sg[t * 4 + 2] - sg[t * 4 + 0] + 1.f;
    float gh = sg[t * 4 + 3] - sg[t * 4 + 1] + 1.f;
    sga[t] = (gw > 0.f && gh > 0.f) ? gw * gh : 0.f;
    skey[t] = 0u;
  }
  __syncthreads();
  float im_w = (float)imw[0], im_h = (float)imh[0];
  int g = t & 63, sub = t >> 6;
  int base = blockIdx.x * 256;
  float gx1 = sg[g * 4 + 0], gy1 = sg[g * 4 + 1];
  float gx2 = sg[g * 4 + 2], gy2 = sg[g * 4 + 3];
  float ga = sga[g];
  float m = -2.f;
  for (int j = sub; j < 256; j += 4) {
    int i = base + j;
    float x1, y1, x2, y2;
    anchor_coords(i, x1, y1, x2, y2);
    bool valid = (x1 >= 0.f) && (y1 >= 0.f) && (x2 < im_w) && (y2 < im_h);
    float aw = x2 - x1 + 1.f, ah = y2 - y1 + 1.f;
    float o = iou_box(x1, y1, x2, y2, aw * ah, gx1, gy1, gx2, gy2, ga);
    m = fmaxf(m, valid ? o : -1.f);
  }
  atomicMax(&skey[g], f2key(m));
  __syncthreads();
  if (t < NG) atomicMax(&gmaxKey[t], skey[t]);
}

// ---------------------------------------------------------------------------
// Labels + argmax + WMMA population count.
// Block = 256 threads (8 wave32) covers 4096 anchors; each wave covers 512:
// lane collects 16 fg/bg flags into a v16h A-matrix, multiplies by all-ones B
// via v_wmma_f32_16x16x32_f16 => D sums the flags (sum(D) = 16*count, exact).
// ---------------------------------------------------------------------------
__global__ void k_labels(const float* __restrict__ gt, const int* __restrict__ imw,
                         const int* __restrict__ imh,
                         const unsigned* __restrict__ gmaxKey,
                         signed char* __restrict__ lblArr,
                         unsigned char* __restrict__ argArr,
                         int* __restrict__ blockFg, int* __restrict__ blockBg) {
  __shared__ float sg[NG * 4];
  __shared__ float sga[NG];
  __shared__ float sgm[NG];
  __shared__ float wsumF[8], wsumB[8];
  int t = threadIdx.x;
  __builtin_prefetch(gt, 0, 0);                 // global_prefetch_b8
  stage_gt_async(gt, sg, t);
  __syncthreads();
  if (t < NG) {
    float gw = sg[t * 4 + 2] - sg[t * 4 + 0] + 1.f;
    float gh = sg[t * 4 + 3] - sg[t * 4 + 1] + 1.f;
    sga[t] = (gw > 0.f && gh > 0.f) ? gw * gh : 0.f;
    sgm[t] = key2f(gmaxKey[t]);
  }
  __syncthreads();
  float im_w = (float)imw[0], im_h = (float)imh[0];
  int wave = t >> 5, lane = t & 31;
  int base = blockIdx.x * 4096 + wave * 512;

  v16h fgv, bgv;
  for (int j = 0; j < 16; ++j) {
    int i = base + j * 32 + lane;
    float x1, y1, x2, y2;
    anchor_coords(i, x1, y1, x2, y2);
    bool valid = (x1 >= 0.f) && (y1 >= 0.f) && (x2 < im_w) && (y2 < im_h);
    float aw = x2 - x1 + 1.f, ah = y2 - y1 + 1.f;
    float aarea = aw * ah;
    float amax = -2.f;
    int aarg = 0;
    bool gtfg = false;
    for (int g = 0; g < NG; ++g) {
      float o = iou_box(x1, y1, x2, y2, aarea,
                        sg[g * 4 + 0], sg[g * 4 + 1], sg[g * 4 + 2], sg[g * 4 + 3],
                        sga[g]);
      if (o > amax) { amax = o; aarg = g; }      // first-max semantics (argmax)
      if (valid && (o == sgm[g])) gtfg = true;   // best-anchor-per-gt
    }
    int lb = 2;                                   // IGN
    if (valid && amax < 0.3f) lb = 0;             // BG
    if (gtfg) lb = 1;                             // FG (gt best)
    if (valid && amax >= 0.7f) lb = 1;            // FG (high overlap)
    lblArr[i] = (signed char)lb;
    argArr[i] = (unsigned char)aarg;
    fgv[j] = (lb == 1) ? (_Float16)1.0f : (_Float16)0.0f;
    bgv[j] = (lb == 0) ? (_Float16)1.0f : (_Float16)0.0f;
  }

  v16h ones;
  for (int j = 0; j < 16; ++j) ones[j] = (_Float16)1.0f;
  v8f cf = {}, cb = {};
  // (neg_a, A, neg_b, B, c_mod, C, reuse_a, reuse_b)  -> v_wmma_f32_16x16x32_f16
  cf = __builtin_amdgcn_wmma_f32_16x16x32_f16(false, fgv, false, ones,
                                              (short)0, cf, false, false);
  cb = __builtin_amdgcn_wmma_f32_16x16x32_f16(false, bgv, false, ones,
                                              (short)0, cb, false, false);
  float sf = 0.f, sb = 0.f;
  for (int r = 0; r < 8; ++r) { sf += cf[r]; sb += cb[r]; }
  for (int off = 16; off; off >>= 1) {
    sf += __shfl_xor(sf, off, 32);
    sb += __shfl_xor(sb, off, 32);
  }
  if (lane == 0) { wsumF[wave] = sf; wsumB[wave] = sb; }
  __syncthreads();
  if (t == 0) {
    float tf = 0.f, tb = 0.f;
    for (int w2 = 0; w2 < 8; ++w2) { tf += wsumF[w2]; tb += wsumB[w2]; }
    blockFg[blockIdx.x] = (int)(tf * (1.f / 16.f) + 0.5f);   // sum(D) = 16*count
    blockBg[blockIdx.x] = (int)(tb * (1.f / 16.f) + 0.5f);
  }
}

// Tiny exclusive scan of 144 per-block counts + num_ni.
__global__ void k_scan(const int* __restrict__ blockFg, const int* __restrict__ blockBg,
                       int* __restrict__ baseFg, int* __restrict__ baseBg,
                       float* __restrict__ invNumNi) {
  if (threadIdx.x == 0 && blockIdx.x == 0) {
    int af = 0, ab = 0;
    for (int b = 0; b < LBL_BLOCKS; ++b) {
      baseFg[b] = af; af += blockFg[b];
      baseBg[b] = ab; ab += blockBg[b];
    }
    int kept = (af < CAP ? af : CAP) + (ab < CAP ? ab : CAP);
    invNumNi[0] = 1.0f / (float)kept;
  }
}

// ---------------------------------------------------------------------------
// Deterministic subsample (keep first CAP fg/bg in anchor-index order, ranks
// via wave32 ballot prefix), regression targets, weights, transposed writes.
// Block = 256 threads walks 16 chunks of 256 anchors to keep ranks in order.
// ---------------------------------------------------------------------------
__global__ void k_out(const float* __restrict__ gt, const int* __restrict__ imw,
                      const int* __restrict__ imh,
                      const signed char* __restrict__ lblArr,
                      const unsigned char* __restrict__ argArr,
                      const int* __restrict__ baseFg, const int* __restrict__ baseBg,
                      const float* __restrict__ invNumNi,
                      int* __restrict__ outLbl, float* __restrict__ outAdj,
                      float* __restrict__ outWts) {
  __shared__ float sg[NG * 4];
  __shared__ int wcf[8], wcb[8];
  __shared__ int run[2];
  int t = threadIdx.x;
  __builtin_prefetch(lblArr + (size_t)blockIdx.x * 4096, 0, 0);
  stage_gt_async(gt, sg, t);
  if (t == 0) { run[0] = baseFg[blockIdx.x]; run[1] = baseBg[blockIdx.x]; }
  __syncthreads();
  float inv = invNumNi[0];
  float im_w = (float)imw[0], im_h = (float)imh[0];
  int wave = t >> 5, lane = t & 31;
  unsigned long long ltmask = (1ull << lane) - 1ull;

  for (int c = 0; c < 16; ++c) {
    int i = blockIdx.x * 4096 + c * 256 + t;
    int lb = (int)lblArr[i];
    bool fg = (lb == 1), bg = (lb == 0);
    unsigned long long bf = __ballot(fg);
    unsigned long long bb = __ballot(bg);
    int pf = __popcll(bf & ltmask);
    int pb = __popcll(bb & ltmask);
    if (lane == 0) { wcf[wave] = __popcll(bf); wcb[wave] = __popcll(bb); }
    __syncthreads();
    int of = 0, ob = 0;
    for (int w2 = 0; w2 < wave; ++w2) { of += wcf[w2]; ob += wcb[w2]; }
    if (fg && (run[0] + of + pf) >= CAP) lb = 2;
    if (bg && (run[1] + ob + pb) >= CAP) lb = 2;

    // labels_out: [a, h, w]
    int a = i % NA, pos = i / NA, w = pos % WW, h = pos / WW;
    int ohw = h * WW + w;
    outLbl[a * (HH * WW) + ohw] = lb;

    // bbox regression targets vs matched gt
    float x1, y1, x2, y2;
    anchor_coords(i, x1, y1, x2, y2);
    bool valid = (x1 >= 0.f) && (y1 >= 0.f) && (x2 < im_w) && (y2 < im_h);
    float aw = x2 - x1 + 1.f, ah = y2 - y1 + 1.f;
    int g = (int)argArr[i];
    float gx1 = sg[g * 4 + 0], gy1 = sg[g * 4 + 1];
    float gx2 = sg[g * 4 + 2], gy2 = sg[g * 4 + 3];
    float gwm = gx2 - gx1 + 1.f, ghm = gy2 - gy1 + 1.f;
    float ax = (x2 + x1) * 0.5f, ay = (y2 + y1) * 0.5f;
    float gx = (gx2 + gx1) * 0.5f, gy = (gy2 + gy1) * 0.5f;
    float a0 = (gx - ax) / aw, a1 = (gy - ay) / ah;
    float a2 = logf(gwm / aw), a3 = logf(ghm / ah);
    if (!valid) { a0 = a1 = a2 = a3 = 0.f; }
    float wt = (lb == 1) ? inv : 0.f;
    int cb4 = 4 * a;
    outAdj[(cb4 + 0) * (HH * WW) + ohw] = a0;
    outAdj[(cb4 + 1) * (HH * WW) + ohw] = a1;
    outAdj[(cb4 + 2) * (HH * WW) + ohw] = a2;
    outAdj[(cb4 + 3) * (HH * WW) + ohw] = a3;
    outWts[(cb4 + 0) * (HH * WW) + ohw] = wt;
    outWts[(cb4 + 1) * (HH * WW) + ohw] = wt;
    outWts[(cb4 + 2) * (HH * WW) + ohw] = wt;
    outWts[(cb4 + 3) * (HH * WW) + ohw] = wt;

    __syncthreads();
    if (t == 0) {
      int tf = 0, tb = 0;
      for (int w2 = 0; w2 < 8; ++w2) { tf += wcf[w2]; tb += wcb[w2]; }
      run[0] += tf; run[1] += tb;
    }
    __syncthreads();
  }
}

extern "C" void kernel_launch(void* const* d_in, const int* in_sizes, int n_in,
                              void* d_out, int out_size, void* d_ws, size_t ws_size,
                              hipStream_t stream) {
  const float* gt  = (const float*)d_in[1];   // [64,4]
  const int*   imw = (const int*)d_in[2];     // scalar
  const int*   imh = (const int*)d_in[3];     // scalar
  (void)d_in[0]; (void)in_sizes; (void)n_in; (void)out_size; (void)ws_size;

  // workspace carve-up (~1.19 MB)
  char* ws = (char*)d_ws;
  size_t off = 0;
  unsigned* gmaxKey = (unsigned*)(ws + off); off += NG * sizeof(unsigned);
  off = (off + 255) & ~(size_t)255;
  signed char* lblArr = (signed char*)(ws + off); off += NANCH;
  unsigned char* argArr = (unsigned char*)(ws + off); off += NANCH;
  off = (off + 255) & ~(size_t)255;
  int* blockFg = (int*)(ws + off); off += LBL_BLOCKS * sizeof(int);
  int* blockBg = (int*)(ws + off); off += LBL_BLOCKS * sizeof(int);
  int* baseFg  = (int*)(ws + off); off += LBL_BLOCKS * sizeof(int);
  int* baseBg  = (int*)(ws + off); off += LBL_BLOCKS * sizeof(int);
  float* invNumNi = (float*)(ws + off); off += sizeof(float);

  // outputs: labels int32 [1,1,9*256,256], adj f32 [1,36,256,256], wts f32 same
  int*   outLbl = (int*)d_out;
  float* outAdj = (float*)d_out + NANCH;
  float* outWts = outAdj + 36 * HH * WW;

  k_init  <<<1, 64, 0, stream>>>(gmaxKey);
  k_gmax  <<<NANCH / 256, 256, 0, stream>>>(gt, imw, imh, gmaxKey);
  k_labels<<<LBL_BLOCKS, 256, 0, stream>>>(gt, imw, imh, gmaxKey,
                                           lblArr, argArr, blockFg, blockBg);
  k_scan  <<<1, 32, 0, stream>>>(blockFg, blockBg, baseFg, baseBg, invNumNi);
  k_out   <<<LBL_BLOCKS, 256, 0, stream>>>(gt, imw, imh, lblArr, argArr,
                                           baseFg, baseBg, invNumNi,
                                           outLbl, outAdj, outWts);
}
